// LSTMNeuralHawkesProcess_3410204033609
// MI455X (gfx1250) — matrix operand order repaired and checked by reading, not compiled, not run on hardware
//
#include <hip/hip_runtime.h>

#define BB   32
#define TT   2048
#define TP1  2049
#define HH   256
#define NG   7
#define NWG  4
#define HS   64            // hidden units per workgroup
#define NTH  256
#define WROW (NG*HH)       // 1792, W row stride

typedef __attribute__((ext_vector_type(16))) _Float16 v16h;
typedef __attribute__((ext_vector_type(8)))  float    v8f;

union HPack { struct { uint4 lo; uint4 hi; } u; v16h v; };

static __device__ __forceinline__ unsigned short f2h(float f) {
    _Float16 h = (_Float16)f;
    return __builtin_bit_cast(unsigned short, h);
}

// ---- hardware transcendentals (TRANS32 ops co-execute with WMMA/XDL) ------
#define LOG2E 1.4426950408889634f
#define LN2   0.6931471805599453f

static __device__ __forceinline__ float fexp(float x) {      // e^x
    return __builtin_amdgcn_exp2f(x * LOG2E);
}
static __device__ __forceinline__ float frcp(float x) {
    return __builtin_amdgcn_rcpf(x);
}
static __device__ __forceinline__ float fsigm(float x) {     // 1/(1+e^-x)
    return frcp(1.0f + __builtin_amdgcn_exp2f(-LOG2E * x));
}
static __device__ __forceinline__ float ftanh(float x) {     // 1 - 2/(e^2x+1)
#if __has_builtin(__builtin_amdgcn_tanhf)
    return __builtin_amdgcn_tanhf(x);
#else
    return 1.0f - 2.0f * frcp(1.0f + __builtin_amdgcn_exp2f(2.0f * LOG2E * x));
#endif
}
static __device__ __forceinline__ float fsoftplus(float x) { // log1p(e^x), stable
    float e = __builtin_amdgcn_exp2f(-LOG2E * fabsf(x));     // in (0,1]
    return fmaxf(x, 0.0f) + LN2 * __builtin_amdgcn_logf(1.0f + e);
}

// ---------------------------------------------------------------------------
// Init kernel: build delta-t (into ws transposed + dt output region), zero the
// t=0 rows of the six (B,T+1,H) outputs, zero both h ping-pong buffers, reset
// the inter-WG barrier counter.
// ---------------------------------------------------------------------------
__global__ __launch_bounds__(256, 1)
void ctlstm_init(const float* __restrict__ pads, const int* __restrict__ lens,
                 float* __restrict__ out, float* __restrict__ ws)
{
    const size_t S  = (size_t)BB * TP1 * HH;
    const size_t DT = (size_t)BB * TP1;
    float*          dtT = ws;                                  // (T+1, B)
    unsigned short* hb  = (unsigned short*)(ws + DT);          // 2 x (B*H) f16
    unsigned*       bar = (unsigned*)(hb + 2 * BB * HH);
    float*          dto = out + 2 * S;

    int idx = blockIdx.x * 256 + threadIdx.x;
    if (idx < (int)DT) {
        int b = idx / TP1, s = idx % TP1;
        int L = lens[b];
        float v;
        if (s < L)       v = (s == 0) ? pads[(size_t)b*TT]
                                      : pads[(size_t)b*TT + s] - pads[(size_t)b*TT + s - 1];
        else if (s == L) v = 100.0f - pads[(size_t)b*TT + (L - 1)];
        else             v = -1.0f;
        dtT[(size_t)s * BB + b]   = v;
        dto[(size_t)b * TP1 + s]  = v;
    }
    if (idx < BB * HH) {                          // zero t=0 rows (6 tensors)
        int b = idx >> 8, k = idx & 255;
        size_t o = ((size_t)b * TP1) * HH + k;
        out[o]              = 0.0f;               // befores
        out[S + o]          = 0.0f;               // afters
        out[2*S + DT + o]   = 0.0f;               // cs
        out[3*S + DT + o]   = 0.0f;               // cbars
        out[4*S + DT + o]   = 0.0f;               // os
        out[5*S + DT + o]   = 0.0f;               // decays
    }
    if (idx < (2 * BB * HH) / 2) ((unsigned*)hb)[idx] = 0u;   // zero both h bufs
    if (idx == 0) { bar[0] = 0u; }
}

// ---------------------------------------------------------------------------
// Main persistent kernel: 4 WGs x 256 threads. Each WG owns a 64-hidden slice
// (all 7 gates => 448 W columns) cached in LDS as f16, pre-swizzled to the
// WMMA-B lane layout. LDS blocks are grouped (hc, g, ks) so each wave's 56
// tiles live in one contiguous 56KB region -> one DS base address VGPR, all
// tile loads via 16-bit immediate offsets (no address spills).
// ---------------------------------------------------------------------------
__global__ __launch_bounds__(256, 1)
void ctlstm_main(const float* __restrict__ Wm, const float* __restrict__ bias,
                 float* __restrict__ out, float* __restrict__ ws)
{
    __shared__ uint4 Wlds[224 * 64];   // 224 blocks * 1KB = 224 KB (f16 slice)

    const size_t S  = (size_t)BB * TP1 * HH;
    const size_t DT = (size_t)BB * TP1;
    float*          dtT = ws;
    unsigned short* hb0 = (unsigned short*)(ws + DT);
    unsigned short* hb1 = hb0 + BB * HH;
    unsigned*       bar = (unsigned*)(hb0 + 2 * BB * HH);

    const int tid = threadIdx.x;
    const int wg  = blockIdx.x;          // 0..3
    const int l   = tid & 31;            // lane
    const int w   = tid >> 5;            // wave 0..7
    const int m   = w & 1;               // batch half (M tile)
    const int hc  = w >> 1;              // hidden 16-chunk 0..3
    const int kl  = hc * 16 + (l & 15);  // local hidden 0..63
    const int kg  = wg * HS + kl;        // global hidden 0..255
    const int vb  = m * 16 + (l >> 4) * 8;   // first batch of this lane's 8 cells
    const int c0  = (l >> 4) * 8;        // A-matrix K sub-offset for this lane
    const int hrow = m * 16 + (l & 15);  // A-matrix source row (batch)

    // ---- stage W slice into LDS, f16, WMMA-B swizzled ----------------------
    // block layout: blk = (hc*7 + g)*8 + ks, 1KB each. Within a block, lane ll
    // element e <- W[1 + ks*32 + (e/8)*16 + (ll/16)*8 + (e%8)][g*256 + wg*64 + hc*16 + (ll%16)]
    unsigned short* Wh = (unsigned short*)Wlds;
    for (int idx = tid; idx < 224 * 512; idx += NTH) {
        int e   = idx & 15;
        int ll  = (idx >> 4) & 31;
        int blk = idx >> 9;              // 0..223
        int ks  = blk & 7;
        int q   = blk >> 3;              // 0..27
        int g   = q % 7;
        int hcb = q / 7;
        int row = 1 + ks * 32 + (e >> 3) * 16 + (ll >> 4) * 8 + (e & 7);
        int col = g * HH + wg * HS + hcb * 16 + (ll & 15);
        Wh[blk * 512 + ll * 16 + e] = f2h(Wm[(size_t)row * WROW + col]);
    }

    // per-lane x-row weight and bias for each gate (constant over time)
    float Wx[NG], bl[NG];
    #pragma unroll
    for (int g = 0; g < NG; ++g) {
        Wx[g] = Wm[g * HH + kg];         // row 0 of W
        bl[g] = bias[g * HH + kg];
    }
    __syncthreads();

    // this wave's contiguous 56KB B-region: single base, immediate offsets
    const uint4* Wb = (const uint4*)((const char*)Wlds + (hc * 57344 + l * 32));

    // persistent cell state: 8 cells per lane (batch vb..vb+7, hidden kg)
    float cst[8], cbst[8];
    #pragma unroll
    for (int v = 0; v < 8; ++v) { cst[v] = 0.0f; cbst[v] = 0.0f; }

    float* befo = out;
    float* afto = out + S;
    float* cso  = out + 2 * S + DT;
    float* cbso = out + 3 * S + DT;
    float* oso  = out + 4 * S + DT;
    float* dso  = out + 5 * S + DT;

    int ping = 0;

    for (int s = 0; s < TT; ++s) {
        const unsigned short* hin  = ping ? hb1 : hb0;
        unsigned short*       hout = ping ? hb0 : hb1;

        // dt(s) and dt(s+1) for this lane's 8 batches (32B contiguous, L2 hot)
        const float4* xr = (const float4*)(dtT + (size_t)s       * BB + vb);
        const float4* dr = (const float4*)(dtT + (size_t)(s + 1) * BB + vb);
        float4 xlo = xr[0], xhi = xr[1];
        float4 dlo = dr[0], dhi = dr[1];
        float xv[8] = {xlo.x, xlo.y, xlo.z, xlo.w, xhi.x, xhi.y, xhi.z, xhi.w};
        float dv[8] = {dlo.x, dlo.y, dlo.z, dlo.w, dhi.x, dhi.y, dhi.z, dhi.w};

        v8f acc[NG];
        #pragma unroll
        for (int g = 0; g < NG; ++g) acc[g] = (v8f){};

        // ---- 8 K-steps x 7 gate tiles of v_wmma_f32_16x16x32_f16 ----------
        const uint4* hrow4 = (const uint4*)(hin + (size_t)hrow * HH);
        #pragma unroll
        for (int ks = 0; ks < 8; ++ks) {
            HPack a;                                   // A: 16x32 f16, ISA layout
            a.u.lo = hrow4[(ks * 32 + c0) >> 3];
            a.u.hi = hrow4[(ks * 32 + c0 + 16) >> 3];
            #pragma unroll
            for (int g = 0; g < NG; ++g) {
                HPack b;                               // B: 32x16 f16 from LDS
                b.u.lo = Wb[(g * 8 + ks) * 64];        // immediate DS offsets
                b.u.hi = Wb[(g * 8 + ks) * 64 + 1];
                acc[g] = __builtin_amdgcn_wmma_f32_16x16x32_f16(
                    false, a.v, false, b.v, (short)0, acc[g], false, false);
            }
        }

        // ---- elementwise gate math, all in registers ----------------------
        #pragma unroll
        for (int v = 0; v < 8; ++v) {
            float x  = xv[v];
            float gi = acc[0][v] + x * Wx[0] + bl[0];
            float gf = acc[1][v] + x * Wx[1] + bl[1];
            float gz = acc[2][v] + x * Wx[2] + bl[2];
            float go = acc[3][v] + x * Wx[3] + bl[3];
            float gb = acc[4][v] + x * Wx[4] + bl[4];
            float gc = acc[5][v] + x * Wx[5] + bl[5];
            float gd = acc[6][v] + x * Wx[6] + bl[6];

            float i_  = fsigm(gi);
            float f_  = fsigm(gf);
            float o_  = fsigm(go);
            float ib_ = fsigm(gb);
            float fb_ = fsigm(gc);
            float z_  = ftanh(gz);
            float d_  = fsoftplus(gd);

            float cn   = f_ * cst[v] + i_ * z_;
            float cbn  = fb_ * cbst[v] + ib_ * z_;
            float aft  = o_ * ftanh(cn);
            float cdec = cbn + (cn - cbn) * fexp(-d_ * dv[v]);
            float hn   = o_ * ftanh(cdec);
            cst[v]  = cdec;
            cbst[v] = cbn;

            size_t oidx = ((size_t)(vb + v) * TP1 + (s + 1)) * HH + kg;
            __builtin_nontemporal_store(hn,  befo + oidx);
            __builtin_nontemporal_store(aft, afto + oidx);
            __builtin_nontemporal_store(cn,  cso  + oidx);
            __builtin_nontemporal_store(cbn, cbso + oidx);
            __builtin_nontemporal_store(o_,  oso  + oidx);
            __builtin_nontemporal_store(d_,  dso  + oidx);
            hout[(size_t)(vb + v) * HH + kg] = f2h(hn);    // next-step h (f16)
        }

        // ---- 4-WG global barrier -----------------------------------------
        // Every thread releases its own stores to device scope first, so the
        // leader's signal publishes the whole workgroup's h updates.
        __threadfence();
        __syncthreads();
        if (tid == 0) {
            unsigned target = (unsigned)(s + 1) * NWG;
            __hip_atomic_fetch_add(&bar[0], 1u, __ATOMIC_ACQ_REL,
                                   __HIP_MEMORY_SCOPE_AGENT);
            while (__hip_atomic_load(&bar[0], __ATOMIC_ACQUIRE,
                                     __HIP_MEMORY_SCOPE_AGENT) < target)
                __builtin_amdgcn_s_sleep(1);
        }
        __syncthreads();
        __threadfence();   // acquire: invalidate L0 so fresh h is visible
        ping ^= 1;
    }
}

extern "C" void kernel_launch(void* const* d_in, const int* in_sizes, int n_in,
                              void* d_out, int out_size, void* d_ws, size_t ws_size,
                              hipStream_t stream)
{
    const float* pads = (const float*)d_in[0];   // seq_pads (B,T,1) f32
    const int*   lens = (const int*)  d_in[1];   // seq_lens (B,)    i32
    const float* Wm   = (const float*)d_in[2];   // W (257, 1792)    f32
    const float* bias = (const float*)d_in[3];   // b (1792,)        f32
    float* out = (float*)d_out;
    float* ws  = (float*)d_ws;

    ctlstm_init<<<dim3((BB * TP1 + 255) / 256), dim3(256), 0, stream>>>(pads, lens, out, ws);
    ctlstm_main<<<dim3(NWG), dim3(NTH), 0, stream>>>(Wm, bias, out, ws);
}
